// TokenMerging_27831388078641
// MI455X (gfx1250) — matrix operand order repaired and checked by reading, not compiled, not run on hardware
//
#include <hip/hip_runtime.h>

// ---------------------------------------------------------------------------
// ToMe token merging for MI455X (gfx1250, wave32, WMMA).
// B=8, T=4096, C=1024, Cm=64, r=1024.
// scores GEMM done with v_wmma_f32_16x16x32_f16 using hi+lo f16 split
// (error-compensated, ~fp32-accurate ranking).
// ---------------------------------------------------------------------------

typedef __attribute__((ext_vector_type(16))) _Float16 v16h;
typedef __attribute__((ext_vector_type(8)))  _Float16 v8h;
typedef __attribute__((ext_vector_type(8)))  float    v8f;

#define NB   8      // batches
#define NT   4096   // tokens
#define HT   2048   // tokens/2 (a and b token counts)
#define NC   1024   // channels of x
#define MC   64     // channels of metric
#define RMRG 1024   // r (merged count)

// ---------------------------------------------------------------------------
// Stage 1: normalize metric rows, split into hi/lo f16 planes, deinterleave
// even tokens -> a planes, odd tokens -> b planes.  Layout: [b][t/2][64].
// ---------------------------------------------------------------------------
__global__ __launch_bounds__(256) void normalize_kernel(
    const float* __restrict__ metric,
    _Float16* __restrict__ aHi, _Float16* __restrict__ aLo,
    _Float16* __restrict__ bHi, _Float16* __restrict__ bLo)
{
    const int wave = threadIdx.x >> 5;
    const int lane = threadIdx.x & 31;
    const int row  = blockIdx.x * 8 + wave;          // 0 .. NB*NT-1
    const int b    = row >> 12;                      // row / 4096
    const int t    = row & 4095;

    const float* mp = metric + (size_t)row * MC;
    float v0 = mp[lane];
    float v1 = mp[lane + 32];
    float ss = v0 * v0 + v1 * v1;
#pragma unroll
    for (int m = 1; m < 32; m <<= 1) ss += __shfl_xor(ss, m, 32);
    const float inv = 1.0f / sqrtf(ss);
    const float n0 = v0 * inv, n1 = v1 * inv;

    const _Float16 h0 = (_Float16)n0;
    const _Float16 h1 = (_Float16)n1;
    const _Float16 l0 = (_Float16)(n0 - (float)h0);
    const _Float16 l1 = (_Float16)(n1 - (float)h1);

    const size_t dbase = ((size_t)b * HT + (size_t)(t >> 1)) * MC;
    _Float16* dh = (t & 1) ? bHi : aHi;
    _Float16* dl = (t & 1) ? bLo : aLo;
    dh[dbase + lane]      = h0;
    dh[dbase + lane + 32] = h1;
    dl[dbase + lane]      = l0;
    dl[dbase + lane + 32] = l1;
}

// Build a 16x32 f16 A-fragment (ISA 7.12.2 layout): per lane, halves
// [k0..k0+7] go to elements 0..7 and [k0+16..k0+23] to elements 8..15.
static __device__ __forceinline__ v16h load_afrag(const _Float16* p, int k0)
{
    v8h lo = *(const v8h*)(p + k0);
    v8h hi = *(const v8h*)(p + k0 + 16);
    v16h r;
#pragma unroll
    for (int e = 0; e < 8; ++e) { r[e] = lo[e]; r[8 + e] = hi[e]; }
    return r;
}

// ---------------------------------------------------------------------------
// Stage 2: scores = a . b^T  with per-row running max/argmax.
// Block = 128 threads (4 waves).  Block handles one 16-row A tile of one
// batch; wave w covers column tiles [w*32, w*32+32).  6 WMMAs per tile:
// AhBh + AhBl + AlBh accumulated in one f32 accumulator (K = 64 in 2 steps).
// ---------------------------------------------------------------------------
__global__ __launch_bounds__(128) void scores_argmax_kernel(
    const _Float16* __restrict__ aHi, const _Float16* __restrict__ aLo,
    const _Float16* __restrict__ bHi, const _Float16* __restrict__ bLo,
    float* __restrict__ node_max, int* __restrict__ node_idx)
{
    const int batch = blockIdx.x >> 7;      // 128 tiles per batch
    const int tileM = blockIdx.x & 127;
    const int lane  = threadIdx.x & 31;
    const int wave  = threadIdx.x >> 5;

    const size_t bb  = (size_t)batch * HT * MC;
    const int mrow   = tileM * 16 + (lane & 15);
    const int k0     = (lane & 16) ? 8 : 0;    // A per-lane K base
    const int kb     = (lane & 16) ? 8 : 0;    // B per-lane K base

    const _Float16* ah = aHi + bb + (size_t)mrow * MC;
    const _Float16* al = aLo + bb + (size_t)mrow * MC;
    const v16h ah0 = load_afrag(ah, k0);
    const v16h ah1 = load_afrag(ah, k0 + 32);
    const v16h al0 = load_afrag(al, k0);
    const v16h al1 = load_afrag(al, k0 + 32);

    float rmax[8];
    int   ridx[8];
#pragma unroll
    for (int r = 0; r < 8; ++r) { rmax[r] = -__builtin_inff(); ridx[r] = 0; }

    for (int it = 0; it < 32; ++it) {
        const int col = (wave * 32 + it) * 16 + (lane & 15);
        const _Float16* bh = bHi + bb + (size_t)col * MC;
        const _Float16* bl = bLo + bb + (size_t)col * MC;
        const v16h bh0 = load_afrag(bh, kb);
        const v16h bh1 = load_afrag(bh, kb + 32);
        const v16h bl0 = load_afrag(bl, kb);
        const v16h bl1 = load_afrag(bl, kb + 32);

        v8f c = {};
        c = __builtin_amdgcn_wmma_f32_16x16x32_f16(false, ah0, false, bh0, (short)0, c, false, false);
        c = __builtin_amdgcn_wmma_f32_16x16x32_f16(false, ah1, false, bh1, (short)0, c, false, false);
        c = __builtin_amdgcn_wmma_f32_16x16x32_f16(false, ah0, false, bl0, (short)0, c, false, false);
        c = __builtin_amdgcn_wmma_f32_16x16x32_f16(false, ah1, false, bl1, (short)0, c, false, false);
        c = __builtin_amdgcn_wmma_f32_16x16x32_f16(false, al0, false, bh0, (short)0, c, false, false);
        c = __builtin_amdgcn_wmma_f32_16x16x32_f16(false, al1, false, bh1, (short)0, c, false, false);

#pragma unroll
        for (int r = 0; r < 8; ++r)
            if (c[r] > rmax[r]) { rmax[r] = c[r]; ridx[r] = col; }
    }

    // Reduce within each 16-lane half (C layout: lane = N, VGPR r = M or M+8).
#pragma unroll
    for (int m = 1; m <= 8; m <<= 1) {
#pragma unroll
        for (int r = 0; r < 8; ++r) {
            const float om = __shfl_xor(rmax[r], m, 32);
            const int   oi = __shfl_xor(ridx[r], m, 32);
            if (om > rmax[r] || (om == rmax[r] && oi < ridx[r])) {
                rmax[r] = om; ridx[r] = oi;
            }
        }
    }

    __shared__ float smax[4][16];
    __shared__ int   sidx[4][16];
    if ((lane & 15) == 0) {
        const int mb = (lane >> 4) * 8;   // lane 0 -> M 0..7, lane 16 -> M 8..15
#pragma unroll
        for (int r = 0; r < 8; ++r) { smax[wave][mb + r] = rmax[r]; sidx[wave][mb + r] = ridx[r]; }
    }
    __syncthreads();

    if (threadIdx.x < 16) {
        float bm = smax[0][threadIdx.x];
        int   bi = sidx[0][threadIdx.x];
#pragma unroll
        for (int w = 1; w < 4; ++w) {     // increasing wave = increasing N: strict >
            if (smax[w][threadIdx.x] > bm) { bm = smax[w][threadIdx.x]; bi = sidx[w][threadIdx.x]; }
        }
        const int node = batch * HT + tileM * 16 + threadIdx.x;
        node_max[node] = bm;
        node_idx[node] = bi;
    }
}

// ---------------------------------------------------------------------------
// Stage 3: bitonic sort of 2048 (key desc, idx asc) per batch == stable
// argsort(-node_max).  Also emit packed (dst<<16)|src for the r merged rows.
// ---------------------------------------------------------------------------
__global__ __launch_bounds__(1024) void sort_kernel(
    const float* __restrict__ node_max, const int* __restrict__ node_idx,
    int* __restrict__ edge_idx, unsigned int* __restrict__ packed)
{
    __shared__ float key[HT];
    __shared__ int   idx[HT];
    const int b = blockIdx.x;
    const int t = threadIdx.x;

    key[t]        = node_max[b * HT + t];        idx[t]        = t;
    key[t + 1024] = node_max[b * HT + t + 1024]; idx[t + 1024] = t + 1024;
    __syncthreads();

    for (int k = 2; k <= HT; k <<= 1) {
        for (int j = k >> 1; j > 0; j >>= 1) {
            const int i = ((t & ~(j - 1)) << 1) | (t & (j - 1));
            const int p = i | j;
            const bool dirF = ((i & k) == 0);    // forward ("before") order region
            const float ki = key[i], kp = key[p];
            const int   ii = idx[i], ip = idx[p];
            // "p belongs before i": larger key first, ties by smaller index
            const bool pBefore = (kp > ki) || (kp == ki && ip < ii);
            if (pBefore == dirF) { key[i] = kp; key[p] = ki; idx[i] = ip; idx[p] = ii; }
            __syncthreads();
        }
    }

    edge_idx[b * HT + t]        = idx[t];
    edge_idx[b * HT + t + 1024] = idx[t + 1024];

    // first RMRG entries are src_idx; dst_idx = node_idx[src]
    const int s = idx[t];
    const int d = node_idx[b * HT + s];
    packed[b * RMRG + t] = ((unsigned)d << 16) | (unsigned)s;
}

// ---------------------------------------------------------------------------
// Stage 4a: unmerged rows -> out rows [0, 1024).  Pure gather, float4.
// ---------------------------------------------------------------------------
__global__ __launch_bounds__(256) void merge_unm_kernel(
    const float* __restrict__ x, const int* __restrict__ edge_idx,
    float* __restrict__ out)
{
    const int b = blockIdx.x >> 10;
    const int i = blockIdx.x & 1023;
    const int s = edge_idx[b * HT + RMRG + i];           // unm_idx[i]
    const float4* src = (const float4*)(x + ((size_t)b * NT + (size_t)(2 * s)) * NC);
    float4*       dst = (float4*)(out + ((size_t)b * 3072 + (size_t)i) * NC);
    dst[threadIdx.x] = src[threadIdx.x];
}

// ---------------------------------------------------------------------------
// Stage 4b: dst rows -> out rows [1024, 3072).  Deterministic scatter-add:
// each dst row scans the packed (dst,src) list in fixed i-order (matches the
// JAX scatter order), accumulates matching src rows, divides by count.
// ---------------------------------------------------------------------------
__global__ __launch_bounds__(256) void merge_dst_kernel(
    const float* __restrict__ x, const unsigned int* __restrict__ packed,
    float* __restrict__ out)
{
    __shared__ unsigned int sp[RMRG];
    const int b = blockIdx.x >> 11;
    const int d = blockIdx.x & 2047;
    for (int i = threadIdx.x; i < RMRG; i += 256) sp[i] = packed[b * RMRG + i];
    __syncthreads();

    const float* xb = x + (size_t)b * NT * NC;
    const float4* drow = (const float4*)(xb + (size_t)(2 * d + 1) * NC);
    float4 acc = drow[threadIdx.x];
    float cnt = 1.0f;

    for (int i = 0; i < RMRG; ++i) {
        const unsigned int p = sp[i];
        if ((int)(p >> 16) == d) {
            const int s = (int)(p & 0xFFFFu);
            const float4* srow = (const float4*)(xb + (size_t)(2 * s) * NC);
            const float4 v = srow[threadIdx.x];
            acc.x += v.x; acc.y += v.y; acc.z += v.z; acc.w += v.w;
            cnt += 1.0f;
        }
    }

    acc.x /= cnt; acc.y /= cnt; acc.z /= cnt; acc.w /= cnt;
    float4* dptr = (float4*)(out + ((size_t)b * 3072 + 1024 + (size_t)d) * NC);
    dptr[threadIdx.x] = acc;
}

// ---------------------------------------------------------------------------
extern "C" void kernel_launch(void* const* d_in, const int* in_sizes, int n_in,
                              void* d_out, int out_size, void* d_ws, size_t ws_size,
                              hipStream_t stream)
{
    const float* x      = (const float*)d_in[0];   // [8,4096,1024] f32
    const float* metric = (const float*)d_in[1];   // [8,4096,64]   f32
    float* out          = (float*)d_out;           // [8,3072,1024] f32

    char* ws = (char*)d_ws;
    const size_t planeBytes = (size_t)NB * HT * MC * sizeof(_Float16);  // 2 MiB
    _Float16* aHi = (_Float16*)(ws);
    _Float16* aLo = (_Float16*)(ws + planeBytes);
    _Float16* bHi = (_Float16*)(ws + 2 * planeBytes);
    _Float16* bLo = (_Float16*)(ws + 3 * planeBytes);
    char* p = ws + 4 * planeBytes;
    float*        node_max = (float*)p;                       p += (size_t)NB * HT * sizeof(float);
    int*          node_idx = (int*)p;                         p += (size_t)NB * HT * sizeof(int);
    int*          edge_idx = (int*)p;                         p += (size_t)NB * HT * sizeof(int);
    unsigned int* packed   = (unsigned int*)p;

    // 1) normalize + hi/lo f16 split (8 rows / 256-thread block)
    normalize_kernel<<<dim3(NB * NT / 8), dim3(256), 0, stream>>>(
        metric, aHi, aLo, bHi, bLo);

    // 2) WMMA scores + row max/argmax (one 16-row tile per block, 4 waves)
    scores_argmax_kernel<<<dim3(NB * (HT / 16)), dim3(128), 0, stream>>>(
        aHi, aLo, bHi, bLo, node_max, node_idx);

    // 3) per-batch bitonic argsort + packed (dst,src)
    sort_kernel<<<dim3(NB), dim3(1024), 0, stream>>>(
        node_max, node_idx, edge_idx, packed);

    // 4a) unmerged gather rows
    merge_unm_kernel<<<dim3(NB * RMRG), dim3(256), 0, stream>>>(
        x, edge_idx, out);

    // 4b) dst rows: deterministic scan-accumulate + divide by merge count
    merge_dst_kernel<<<dim3(NB * HT), dim3(256), 0, stream>>>(
        x, packed, out);
}
